// TransformerDecoderBlock_14181982012037
// MI455X (gfx1250) — matrix operand order repaired and checked by reading, not compiled, not run on hardware
//
#include <hip/hip_runtime.h>

// ---------------------------------------------------------------------------
// MI455X (gfx1250) transformer decoder block, bf16 WMMA everywhere.
// wave32; V_WMMA_F32_16X16X32_BF16; A/B frags use contiguous K (16h+s), C/D per CDNA5 ISA.
// GEMM tiles staged with GLOBAL_LOAD_ASYNC_TO_LDS_B128 into double-buffered
// LDS; depth-aware s_wait_asynccnt overlaps next-tile loads with WMMAs.
// Weights pre-transposed once so no per-k-step LDS transpose is needed.
// ---------------------------------------------------------------------------

typedef __attribute__((ext_vector_type(16))) __bf16 v16bf;
typedef __attribute__((ext_vector_type(8)))  __bf16 v8bf;
typedef __attribute__((ext_vector_type(8)))  float  v8f;

union BF16Frag { v16bf v; v8bf h[2]; };

__device__ __forceinline__ unsigned short f2bf(float f) {
  unsigned int u = __float_as_uint(f);
  u += 0x7FFFu + ((u >> 16) & 1u);          // round-to-nearest-even
  return (unsigned short)(u >> 16);
}

__device__ __forceinline__ v8f wmma_bf16(const BF16Frag& a, const BF16Frag& b, v8f c) {
  // (neg_a, A, neg_b, B, c_mod, C, reuse_a, reuse_b)
  return __builtin_amdgcn_wmma_f32_16x16x32_bf16(false, a.v, false, b.v,
                                                 (short)0, c, false, false);
}

// Async copy of 32 contiguous bytes global -> LDS (per lane).
// IOFFSET adds to BOTH lds and global addresses (ISA 08_async_tensor §4.4).
__device__ __forceinline__ void async_copy32B(unsigned lds, const void* g) {
  asm volatile("global_load_async_to_lds_b128 %0, %1, off"
               :: "v"(lds), "v"(g) : "memory");
  asm volatile("global_load_async_to_lds_b128 %0, %1, off offset:16"
               :: "v"(lds), "v"(g) : "memory");
}

constexpr int MTOK = 4096;   // B*N tokens (all GEMM M dims)

// ---------------------------------------------------------------------------
// fp32 -> bf16 conversion (grid-stride)
// ---------------------------------------------------------------------------
__global__ void cvt_bf16_kernel(const float* __restrict__ src,
                                unsigned short* __restrict__ dst, int n) {
  int i = blockIdx.x * blockDim.x + threadIdx.x;
  int stride = gridDim.x * blockDim.x;
  for (; i < n; i += stride) dst[i] = f2bf(src[i]);
}

// ---------------------------------------------------------------------------
// fp32 [K,N] -> bf16 transposed [N,K]  (one-time weight prep, LDS-tiled)
// block (32,8), grid (N/32, K/32)
// ---------------------------------------------------------------------------
__global__ __launch_bounds__(256) void cvt_t_bf16_kernel(
    const float* __restrict__ src, unsigned short* __restrict__ dst,
    int K, int N) {
  __shared__ unsigned short t[32][33];
  const int n0 = blockIdx.x * 32, k0 = blockIdx.y * 32;
  const int tx = threadIdx.x, ty = threadIdx.y;
#pragma unroll
  for (int i = 0; i < 32; i += 8)
    t[ty + i][tx] = f2bf(src[(size_t)(k0 + ty + i) * N + n0 + tx]);  // t[k][n]
  __syncthreads();
#pragma unroll
  for (int i = 0; i < 32; i += 8)
    dst[(size_t)(n0 + ty + i) * K + k0 + tx] = t[tx][ty + i];        // dst[n][k]
}

// ---------------------------------------------------------------------------
// Tiled WMMA GEMM: C[M,N] = act(A[M,K] @ W[K,N] + bias), W given transposed.
// Block tile 128x128, 8 waves (256 thr), each wave computes 32x64.
// A[M,K] and Bt[N,K] tiles async-staged row-major into double-buffered LDS;
// both fragment types then load as 2x ds_load_b128 per lane.
// MODE: 0 = fp32 out, 1 = bf16 out, 2 = bf16 transposed out [N,MTOK].
// ---------------------------------------------------------------------------
template<int N, int K, int MODE, bool RELU>
__global__ __launch_bounds__(256) void gemm_wmma_bf16(
    const unsigned short* __restrict__ A,    // [M,K]
    const unsigned short* __restrict__ Bt,   // [N,K] pre-transposed weights
    const float* __restrict__ bias,
    float* __restrict__ Cf,
    unsigned short* __restrict__ Cb)
{
  constexpr int AP = 40, BP = 40;                 // padded pitches (bank spread)
  constexpr int NK = K / 32;
  __shared__ unsigned short As[2][128 * AP];
  __shared__ unsigned short Bs[2][128 * BP];

  const int tid  = threadIdx.x;
  const int lane = tid & 31;
  const int wave = tid >> 5;
  const int ln   = lane & 15;
  const int lh   = lane >> 4;
  const int wm   = wave & 3;                      // 32-row chunk within tile
  const int wn   = wave >> 2;                     // 64-col chunk within tile
  const int m0   = blockIdx.y * 128;
  const int n0   = blockIdx.x * 128;

  v8f acc[2][4];
#pragma unroll
  for (int i = 0; i < 2; i++)
#pragma unroll
    for (int j = 0; j < 4; j++)
#pragma unroll
      for (int r = 0; r < 8; r++) acc[i][j][r] = 0.f;

  const int arow = tid >> 1;                      // 0..127
  const int acol = (tid & 1) * 16;                // 0 / 16

  const unsigned short* ga  = A  + (size_t)(m0 + arow) * K + acol;
  const unsigned short* gbt = Bt + (size_t)(n0 + arow) * K + acol;
  const unsigned asl[2] = { (unsigned)(uintptr_t)&As[0][arow * AP + acol],
                            (unsigned)(uintptr_t)&As[1][arow * AP + acol] };
  const unsigned bsl[2] = { (unsigned)(uintptr_t)&Bs[0][arow * BP + acol],
                            (unsigned)(uintptr_t)&Bs[1][arow * BP + acol] };

  // prologue: stage tile 0 into buffer 0
  async_copy32B(asl[0], ga);
  async_copy32B(bsl[0], gbt);

  for (int kk = 0; kk < NK; kk++) {
    const int cur = kk & 1;
    if (kk + 1 < NK) {
      // stage next tile into the other buffer, then wait only for tile kk
      // (async loads complete in order; 4 per-thread ops stay outstanding)
      async_copy32B(asl[cur ^ 1], ga  + (kk + 1) * 32);
      async_copy32B(bsl[cur ^ 1], gbt + (kk + 1) * 32);
      asm volatile("s_wait_asynccnt 0x4" ::: "memory");
    } else {
      asm volatile("s_wait_asynccnt 0x0" ::: "memory");
    }
    __syncthreads();

    const unsigned short* Ab = As[cur];
    const unsigned short* Bb = Bs[cur];
    BF16Frag af[2], bfr[4];
#pragma unroll
    for (int mt = 0; mt < 2; mt++) {
      // A frag 16x32: lane ln -> row M=ln; K = 16h..16h+15 contiguous
      const unsigned short* p = &Ab[(wm * 32 + mt * 16 + ln) * AP + 16 * lh];
      af[mt].h[0] = *(const v8bf*)(p);
      af[mt].h[1] = *(const v8bf*)(p + 8);
    }
#pragma unroll
    for (int nt = 0; nt < 4; nt++) {
      // B frag 32x16: lane ln -> col N=ln; K = 16h..16h+15 contiguous
      const unsigned short* p = &Bb[(wn * 64 + nt * 16 + ln) * BP + 16 * lh];
      bfr[nt].h[0] = *(const v8bf*)(p);
      bfr[nt].h[1] = *(const v8bf*)(p + 8);
    }
#pragma unroll
    for (int mt = 0; mt < 2; mt++)
#pragma unroll
      for (int nt = 0; nt < 4; nt++)
        acc[mt][nt] = wmma_bf16(af[mt], bfr[nt], acc[mt][nt]);
    __syncthreads();   // protect buffer re-issued next iteration
  }

  // Epilogue: C/D layout -> lane ln = col, VGPR r -> row r + 8h
#pragma unroll
  for (int mt = 0; mt < 2; mt++) {
    const int mbase = m0 + wm * 32 + mt * 16 + 8 * lh;   // + r
#pragma unroll
    for (int nt = 0; nt < 4; nt++) {
      const int n = n0 + wn * 64 + nt * 16 + ln;
      const float bb = bias[n];
      float v[8];
#pragma unroll
      for (int r = 0; r < 8; r++) {
        float t = acc[mt][nt][r] + bb;
        if (RELU) t = fmaxf(t, 0.f);
        v[r] = t;
      }
      if (MODE == 0) {
        float* p = Cf + (size_t)mbase * N + n;
#pragma unroll
        for (int r = 0; r < 8; r++) p[r * N] = v[r];
      } else if (MODE == 1) {
        unsigned short* p = Cb + (size_t)mbase * N + n;
#pragma unroll
        for (int r = 0; r < 8; r++) p[r * N] = f2bf(v[r]);
      } else {
        // 8 consecutive rows at fixed col -> one 16B store into [N][MTOK]
        uint4 q;
        q.x = (unsigned)f2bf(v[0]) | ((unsigned)f2bf(v[1]) << 16);
        q.y = (unsigned)f2bf(v[2]) | ((unsigned)f2bf(v[3]) << 16);
        q.z = (unsigned)f2bf(v[4]) | ((unsigned)f2bf(v[5]) << 16);
        q.w = (unsigned)f2bf(v[6]) | ((unsigned)f2bf(v[7]) << 16);
        *(uint4*)(&Cb[(size_t)n * MTOK + mbase]) = q;
      }
    }
  }
}

// ---------------------------------------------------------------------------
// Flash-attention core. One wave per (b, h, 16-row q tile).
// Q,K: bf16 [B*N, 512] row-major (head h = cols h*64..h*64+63)
// Vt : bf16 [512, B*N]  (dh-major, i.e. V transposed) -> direct B-frag loads
// Hd : bf16 [B*N, 512] output (concatenated heads)
// Processes keys 32 at a time (two 16x16 score tiles -> one 16x32 P A-frag).
// ---------------------------------------------------------------------------
template<bool CAUSAL>
__global__ __launch_bounds__(256) void attn_core(
    const unsigned short* __restrict__ Q,
    const unsigned short* __restrict__ Kb,
    const unsigned short* __restrict__ Vt,
    unsigned short* __restrict__ Hd)
{
  constexpr int Nseq = 2048, D = 512, PP = 40;
  __shared__ unsigned short Pl[8 * 16 * PP];      // per-wave P staging

  const int lane = threadIdx.x & 31;
  const int wv   = threadIdx.x >> 5;
  const int gw   = blockIdx.x * 8 + wv;           // 2048 waves total
  const int qt   = gw & 127;                      // q tile (16 rows)
  const int h    = (gw >> 7) & 7;
  const int b    = gw >> 10;
  const int ln   = lane & 15;
  const int lh   = lane >> 4;

  unsigned short* Pw = &Pl[wv * 16 * PP];

  BF16Frag aq[2];                                  // Q 16x64 as two A-frags
  {
    const unsigned short* qp = Q + (size_t)(b * Nseq + qt * 16 + ln) * D + h * 64;
#pragma unroll
    for (int c = 0; c < 2; c++) {
      aq[c].h[0] = *(const v8bf*)(qp + c * 32 + 16 * lh);
      aq[c].h[1] = *(const v8bf*)(qp + c * 32 + 16 * lh + 8);
    }
  }

  v8f vzero;
#pragma unroll
  for (int r = 0; r < 8; r++) vzero[r] = 0.f;

  v8f acc[4];
#pragma unroll
  for (int t = 0; t < 4; t++) acc[t] = vzero;
  float mrow[8], lrow[8];
#pragma unroll
  for (int r = 0; r < 8; r++) { mrow[r] = -3.0e38f; lrow[r] = 0.f; }

  const int nsteps = CAUSAL ? (qt / 2 + 1) : (Nseq / 32);

  for (int js = 0; js < nsteps; js++) {
    const int jj = js * 32;
    v8f S[2];
#pragma unroll
    for (int s = 0; s < 2; s++) {                  // score tile: Q @ K^T chunk
      const unsigned short* kp =
          Kb + (size_t)(b * Nseq + jj + s * 16 + ln) * D + h * 64;
      BF16Frag bk0, bk1;
      bk0.h[0] = *(const v8bf*)(kp + 16 * lh);
      bk0.h[1] = *(const v8bf*)(kp + 16 * lh + 8);
      bk1.h[0] = *(const v8bf*)(kp + 32 + 16 * lh);
      bk1.h[1] = *(const v8bf*)(kp + 32 + 16 * lh + 8);
      v8f z = wmma_bf16(aq[0], bk0, vzero);
      S[s]  = wmma_bf16(aq[1], bk1, z);
    }

    float s0[8], s1[8];
#pragma unroll
    for (int r = 0; r < 8; r++) {
      float a0 = S[0][r] * 0.125f;                 // 1/sqrt(64)
      float a1 = S[1][r] * 0.125f;
      if (CAUSAL) {
        const int row = qt * 16 + r + 8 * lh;
        if (jj + ln      > row) a0 = -1.0e30f;
        if (jj + 16 + ln > row) a1 = -1.0e30f;
      }
      s0[r] = a0; s1[r] = a1;
    }

#pragma unroll
    for (int r = 0; r < 8; r++) {                  // online softmax
      float mx = fmaxf(s0[r], s1[r]);
#pragma unroll
      for (int d = 1; d < 16; d <<= 1) mx = fmaxf(mx, __shfl_xor(mx, d, 32));
      const float nm = fmaxf(mrow[r], mx);
      const float sc = __expf(mrow[r] - nm);
      mrow[r] = nm;
      const float p0 = __expf(s0[r] - nm);
      const float p1 = __expf(s1[r] - nm);
      float rs = p0 + p1;
#pragma unroll
      for (int d = 1; d < 16; d <<= 1) rs += __shfl_xor(rs, d, 32);
      lrow[r] = lrow[r] * sc + rs;
#pragma unroll
      for (int t = 0; t < 4; t++) acc[t][r] *= sc;
      Pw[(r + 8 * lh) * PP + ln]      = f2bf(p0);  // C-layout -> LDS
      Pw[(r + 8 * lh) * PP + 16 + ln] = f2bf(p1);
    }
    asm volatile("s_wait_dscnt 0x0" ::: "memory"); // P visible within wave

    BF16Frag pf;                                   // reload P as A-frag 16x32
    {
      const unsigned short* pp = &Pw[ln * PP + 16 * lh];
      pf.h[0] = *(const v8bf*)(pp);
      pf.h[1] = *(const v8bf*)(pp + 8);
    }
#pragma unroll
    for (int t = 0; t < 4; t++) {                  // acc += P @ V(32x64 chunk)
      const unsigned short* vp = Vt +
          (size_t)(h * 64 + t * 16 + ln) * (2 * Nseq) + b * Nseq + jj + 16 * lh;
      BF16Frag bv;
      bv.h[0] = *(const v8bf*)(vp);
      bv.h[1] = *(const v8bf*)(vp + 8);
      acc[t] = wmma_bf16(pf, bv, acc[t]);
    }
  }

#pragma unroll
  for (int t = 0; t < 4; t++)
#pragma unroll
    for (int r = 0; r < 8; r++) {
      const float v = acc[t][r] / lrow[r];
      const int row = b * Nseq + qt * 16 + r + 8 * lh;
      const int col = h * 64 + t * 16 + ln;
      Hd[(size_t)row * D + col] = f2bf(v);
    }
}

// ---------------------------------------------------------------------------
// out = LayerNorm(X + Y) * g + b ; fp32 out (+ optional bf16 mirror)
// One 128-thread block per row, D = 512.
// ---------------------------------------------------------------------------
template<bool WRITE_BF16>
__global__ __launch_bounds__(128) void add_ln(
    const float* __restrict__ X, const float* __restrict__ Y,
    const float* __restrict__ g, const float* __restrict__ bta,
    float* __restrict__ outf, unsigned short* __restrict__ outb)
{
  constexpr int D = 512;
  __shared__ float red[2][4];
  const int row = blockIdx.x;
  const int tid = threadIdx.x;
  const float* x = X + (size_t)row * D;
  const float* y = Y + (size_t)row * D;
  float v[4], s1 = 0.f, s2 = 0.f;
#pragma unroll
  for (int i = 0; i < 4; i++) {
    const float t = x[tid + i * 128] + y[tid + i * 128];
    v[i] = t; s1 += t; s2 += t * t;
  }
#pragma unroll
  for (int d = 1; d < 32; d <<= 1) {
    s1 += __shfl_xor(s1, d, 32);
    s2 += __shfl_xor(s2, d, 32);
  }
  const int wv = tid >> 5;
  if ((tid & 31) == 0) { red[0][wv] = s1; red[1][wv] = s2; }
  __syncthreads();
  s1 = red[0][0] + red[0][1] + red[0][2] + red[0][3];
  s2 = red[1][0] + red[1][1] + red[1][2] + red[1][3];
  const float mu  = s1 / D;
  const float var = s2 / D - mu * mu;
  const float rs  = rsqrtf(var + 1e-6f);
#pragma unroll
  for (int i = 0; i < 4; i++) {
    const int c = tid + i * 128;
    const float o = (v[i] - mu) * rs * g[c] + bta[c];
    outf[(size_t)row * D + c] = o;
    if (WRITE_BF16) outb[(size_t)row * D + c] = f2bf(o);
  }
}

// ---------------------------------------------------------------------------
// Host launcher
// ---------------------------------------------------------------------------
extern "C" void kernel_launch(void* const* d_in, const int* in_sizes, int n_in,
                              void* d_out, int out_size, void* d_ws, size_t ws_size,
                              hipStream_t stream) {
  (void)in_sizes; (void)n_in; (void)out_size; (void)ws_size;
  const int Btok = MTOK, D = 512, FF = 2048;

  const float* inputs = (const float*)d_in[0];
  const float* enc    = (const float*)d_in[1];
  // d_in[2] look_ahead_mask (causal, applied analytically), d_in[3] padding (all-zero)
  const float* W[8]; const float* Bv[8];
  for (int i = 0; i < 8; i++) {
    W[i]  = (const float*)d_in[4 + 2 * i];
    Bv[i] = (const float*)d_in[5 + 2 * i];
  }
  const float* f1w  = (const float*)d_in[20]; const float* f1b  = (const float*)d_in[21];
  const float* f2w  = (const float*)d_in[22]; const float* f2b  = (const float*)d_in[23];
  const float* ln1g = (const float*)d_in[24]; const float* ln1b = (const float*)d_in[25];
  const float* ln2g = (const float*)d_in[26]; const float* ln2b = (const float*)d_in[27];
  const float* ln3g = (const float*)d_in[28]; const float* ln3b = (const float*)d_in[29];

  char* base = (char*)d_ws;
  size_t off = 0;
  auto alloc = [&](size_t bytes) {
    void* p = base + off;
    off = (off + bytes + 255) & ~(size_t)255;
    return p;
  };
  unsigned short* xb    = (unsigned short*)alloc((size_t)Btok * D * 2);
  unsigned short* eb    = (unsigned short*)alloc((size_t)Btok * D * 2);
  unsigned short* wb[8];                      // transposed [N,K]
  for (int i = 0; i < 8; i++) wb[i] = (unsigned short*)alloc((size_t)D * D * 2);
  unsigned short* f1wb  = (unsigned short*)alloc((size_t)D * FF * 2);  // [FF,D]
  unsigned short* f2wb  = (unsigned short*)alloc((size_t)FF * D * 2);  // [D,FF]
  unsigned short* Qb    = (unsigned short*)alloc((size_t)Btok * D * 2);
  unsigned short* Kb    = (unsigned short*)alloc((size_t)Btok * D * 2);
  unsigned short* Vt    = (unsigned short*)alloc((size_t)D * Btok * 2);
  unsigned short* Hd    = (unsigned short*)alloc((size_t)Btok * D * 2);
  float*          tmpf  = (float*)alloc((size_t)Btok * D * 4);
  float*          out1f = (float*)alloc((size_t)Btok * D * 4);
  unsigned short* out1b = (unsigned short*)alloc((size_t)Btok * D * 2);
  float*          out2f = (float*)alloc((size_t)Btok * D * 4);
  unsigned short* out2b = (unsigned short*)alloc((size_t)Btok * D * 2);
  unsigned short* ffh   = (unsigned short*)alloc((size_t)Btok * FF * 2);

  // activations: plain fp32 -> bf16
  cvt_bf16_kernel<<<512, 256, 0, stream>>>(inputs, xb, Btok * D);
  cvt_bf16_kernel<<<512, 256, 0, stream>>>(enc,    eb, Btok * D);
  // weights: fp32 [K,N] -> bf16 transposed [N,K]
  const dim3 tblk(32, 8);
  for (int i = 0; i < 8; i++)
    cvt_t_bf16_kernel<<<dim3(D / 32, D / 32), tblk, 0, stream>>>(W[i], wb[i], D, D);
  cvt_t_bf16_kernel<<<dim3(FF / 32, D / 32), tblk, 0, stream>>>(f1w, f1wb, D, FF);
  cvt_t_bf16_kernel<<<dim3(D / 32, FF / 32), tblk, 0, stream>>>(f2w, f2wb, FF, D);

  const dim3 blk(256);
  const dim3 gP(D / 128, Btok / 128);     // 4 x 32
  const dim3 gF(FF / 128, Btok / 128);    // 16 x 32

  // ---- self attention (causal) ----
  gemm_wmma_bf16<512, 512, 1, false><<<gP, blk, 0, stream>>>(xb, wb[0], Bv[0], nullptr, Qb);
  gemm_wmma_bf16<512, 512, 1, false><<<gP, blk, 0, stream>>>(xb, wb[1], Bv[1], nullptr, Kb);
  gemm_wmma_bf16<512, 512, 2, false><<<gP, blk, 0, stream>>>(xb, wb[2], Bv[2], nullptr, Vt);
  attn_core<true><<<256, 256, 0, stream>>>(Qb, Kb, Vt, Hd);
  gemm_wmma_bf16<512, 512, 0, false><<<gP, blk, 0, stream>>>(Hd, wb[3], Bv[3], tmpf, nullptr);
  add_ln<true><<<Btok, 128, 0, stream>>>(inputs, tmpf, ln1g, ln1b, out1f, out1b);

  // ---- cross attention: q = enc, k = enc, v = out1 (reference quirk) ----
  gemm_wmma_bf16<512, 512, 1, false><<<gP, blk, 0, stream>>>(eb, wb[4], Bv[4], nullptr, Qb);
  gemm_wmma_bf16<512, 512, 1, false><<<gP, blk, 0, stream>>>(eb, wb[5], Bv[5], nullptr, Kb);
  gemm_wmma_bf16<512, 512, 2, false><<<gP, blk, 0, stream>>>(out1b, wb[6], Bv[6], nullptr, Vt);
  attn_core<false><<<256, 256, 0, stream>>>(Qb, Kb, Vt, Hd);
  gemm_wmma_bf16<512, 512, 0, false><<<gP, blk, 0, stream>>>(Hd, wb[7], Bv[7], tmpf, nullptr);
  add_ln<true><<<Btok, 128, 0, stream>>>(out1f, tmpf, ln2g, ln2b, out2f, out2b);

  // ---- FFN ----
  gemm_wmma_bf16<2048, 512, 1, true><<<gF, blk, 0, stream>>>(out2b, f1wb, f1b, nullptr, ffh);
  gemm_wmma_bf16<512, 2048, 0, false><<<gP, blk, 0, stream>>>(ffh, f2wb, f2b, tmpf, nullptr);
  add_ln<false><<<Btok, 128, 0, stream>>>(out2f, tmpf, ln3g, ln3b, (float*)d_out, nullptr);
}
